// EnhancedAttentionLayer_33861522162195
// MI455X (gfx1250) — matrix operand order repaired and checked by reading, not compiled, not run on hardware
//
#include <hip/hip_runtime.h>
#include <cmath>

// ---------------- problem constants ----------------
#define BATCH   4
#define S_LEN   2048
#define H_DIM   1024
#define NH      8
#define HD      128           // H_DIM / NH

typedef __bf16 bf16_t;
typedef __bf16 v16bf __attribute__((ext_vector_type(16)));
typedef __bf16 v8bf  __attribute__((ext_vector_type(8)));
typedef float  v8f   __attribute__((ext_vector_type(8)));

// WMMA wrapper: D = A(16x32 bf16) x B(32x16 bf16) + C(16x16 f32)
__device__ __forceinline__ v8f wmma_bf16(v16bf a, v16bf b, v8f c) {
  return __builtin_amdgcn_wmma_f32_16x16x32_bf16(
      /*neg_a=*/false, a, /*neg_b=*/false, b,
      /*c_mod=*/(short)0, c, /*reuse_a=*/false, /*reuse_b=*/false);
}

// A-fragment loader: base points at row m, k-offset already includes (lane>=16)*16.
// Halves 0..7  come from base[0..7]   (K = koff + 0..7)
// Halves 8..15 come from base[8..15]  (K = koff + 8..15)  -- contiguous k, matches B.
__device__ __forceinline__ v16bf load_a_chunks(const bf16_t* base) {
  v8bf lo = *(const v8bf*)(base);
  v8bf hi = *(const v8bf*)(base + 8);
  v16bf r;
#pragma unroll
  for (int i = 0; i < 8; ++i) { r[i] = lo[i]; r[i + 8] = hi[i]; }
  return r;
}

__device__ __forceinline__ void wait_dscnt0() {
#if __has_builtin(__builtin_amdgcn_s_wait_dscnt)
  __builtin_amdgcn_s_wait_dscnt(0);
#else
  asm volatile("s_wait_dscnt 0x0" ::: "memory");
#endif
}

// ---------------- fp32 -> bf16 conversion ----------------
__global__ void cvt_bf16_kernel(const float* __restrict__ src,
                                bf16_t* __restrict__ dst, int n) {
  int i = blockIdx.x * blockDim.x + threadIdx.x;
  int stride = gridDim.x * blockDim.x;
  for (; i < n; i += stride) dst[i] = (bf16_t)src[i];
}

// ---------------- QKV projection: y = x @ W^T + b ----------------
// X: [M=B*S, H] bf16 row-major, W: [H, H] bf16 row-major (we use W^T => B[k][n] = W[n][k],
// contiguous in k).  out: normal [M,H] (Q,K) or transposed [B][H][S] (V).
__global__ __launch_bounds__(256) void qkv_gemm_kernel(
    const bf16_t* __restrict__ X, const bf16_t* __restrict__ W,
    const float* __restrict__ bias, bf16_t* __restrict__ out, int transpose_out) {
  const int lane = threadIdx.x & 31;
  const int wid  = threadIdx.x >> 5;
  const int tile = blockIdx.x * 8 + wid;        // 32768 tiles total
  const int tm = tile >> 6;                     // M tiles: 512
  const int tn = tile & 63;                     // N tiles: 64
  const int row0 = tm * 16, col0 = tn * 16;
  const int m16 = lane & 15;
  const int hl  = lane >> 4;
  const int koffA = hl * 16, koffB = hl * 16;

  v8f acc = {};
  for (int k0 = 0; k0 < H_DIM; k0 += 32) {
    const bf16_t* abase = X + (size_t)(row0 + m16) * H_DIM + k0 + koffA;
    __builtin_prefetch(abase + 64, 0, 1);       // k-block two ahead
    v16bf a   = load_a_chunks(abase);
    v16bf bfr = *(const v16bf*)(W + (size_t)(col0 + m16) * H_DIM + k0 + koffB);
    acc = wmma_bf16(a, bfr, acc);
  }
  const float bj = bias[col0 + m16];
#pragma unroll
  for (int v = 0; v < 8; ++v) {
    const int r = v + 8 * hl;                   // output row within tile
    const float val = acc[v] + bj;
    const int grow = row0 + r;
    if (transpose_out) {                        // out[b][col][s], S_LEN = 2048
      const int bb = grow >> 11;
      const int s  = grow & (S_LEN - 1);
      out[((size_t)bb * H_DIM + col0 + m16) * S_LEN + s] = (bf16_t)val;
    } else {
      out[(size_t)grow * H_DIM + col0 + m16] = (bf16_t)val;
    }
  }
}

// ---------------- fused flash attention per (b,h) ----------------
// One workgroup (4 waves) owns one (b,h): deterministic accumulation of
// avg_attn column sums + context column sums in LDS, no atomics.
__global__ __launch_bounds__(128) void attn_kernel(
    const bf16_t* __restrict__ Q, const bf16_t* __restrict__ K,
    const bf16_t* __restrict__ Vt, float* __restrict__ avg_out,
    float* __restrict__ ctx_acc) {
  __shared__ float avg_l[4][S_LEN];             // 32 KB: per-wave key column sums
  __shared__ float ctx_l[4][HD];                //  2 KB: per-wave output column sums
  __shared__ __align__(32) bf16_t p_l[4][16 * 32];  // 4 KB: P staging (C->A transpose)

  const int tid  = threadIdx.x;
  const int wid  = tid >> 5;
  const int lane = tid & 31;
  const int b = blockIdx.x / NH;
  const int h = blockIdx.x % NH;

  for (int i = tid; i < 4 * S_LEN; i += 128) (&avg_l[0][0])[i] = 0.f;
  for (int i = tid; i < 4 * HD;    i += 128) (&ctx_l[0][0])[i] = 0.f;
  __syncthreads();

  const bf16_t* Qb = Q  + ((size_t)b * S_LEN) * H_DIM + h * HD;
  const bf16_t* Kb = K  + ((size_t)b * S_LEN) * H_DIM + h * HD;
  const bf16_t* Vb = Vt + ((size_t)b * H_DIM + h * HD) * S_LEN;

  const int m16 = lane & 15;
  const int hl  = lane >> 4;
  const int koffA = hl * 16, koffB = hl * 16;
  const float scale = 0.088388347648318447f;    // 1/sqrt(128)
  bf16_t* pl = &p_l[wid][0];

  for (int qt = wid; qt < S_LEN / 16; qt += 4) {
    const int q0 = qt * 16;
    v16bf qa[4];
#pragma unroll
    for (int kk = 0; kk < 4; ++kk)
      qa[kk] = load_a_chunks(Qb + (size_t)(q0 + m16) * H_DIM + kk * 32 + koffA);

    float mrow[8], lrow[8];
#pragma unroll
    for (int v = 0; v < 8; ++v) { mrow[v] = -1e30f; lrow[v] = 0.f; }

    // ---- pass 1: online max / sum over all keys ----
    for (int j = 0; j < S_LEN; j += 16) {
      v8f s = {};
#pragma unroll
      for (int kk = 0; kk < 4; ++kk) {
        v16bf kf = *(const v16bf*)(Kb + (size_t)(j + m16) * H_DIM + kk * 32 + koffB);
        s = wmma_bf16(qa[kk], kf, s);
      }
#pragma unroll
      for (int v = 0; v < 8; ++v) {
        float x = s[v] * scale;
        float tm = x;
        tm = fmaxf(tm, __shfl_xor(tm, 1, 32));
        tm = fmaxf(tm, __shfl_xor(tm, 2, 32));
        tm = fmaxf(tm, __shfl_xor(tm, 4, 32));
        tm = fmaxf(tm, __shfl_xor(tm, 8, 32));
        const float nm = fmaxf(mrow[v], tm);
        float e = __expf(x - nm);
        e += __shfl_xor(e, 1, 32);
        e += __shfl_xor(e, 2, 32);
        e += __shfl_xor(e, 4, 32);
        e += __shfl_xor(e, 8, 32);
        lrow[v] = lrow[v] * __expf(mrow[v] - nm) + e;
        mrow[v] = nm;
      }
    }
    float rl[8];
#pragma unroll
    for (int v = 0; v < 8; ++v) rl[v] = 1.f / lrow[v];

    v8f o[8];
    {
      v8f z = {};
#pragma unroll
      for (int c = 0; c < 8; ++c) o[c] = z;
    }

    // ---- pass 2: P = exp(s-m)/l ; O += P @ V ; avg += colsum(P) ----
    for (int j = 0; j < S_LEN; j += 32) {
      float csum[2];
#pragma unroll
      for (int t = 0; t < 2; ++t) {
        v8f s = {};
#pragma unroll
        for (int kk = 0; kk < 4; ++kk) {
          v16bf kf = *(const v16bf*)(Kb + (size_t)(j + t * 16 + m16) * H_DIM + kk * 32 + koffB);
          s = wmma_bf16(qa[kk], kf, s);
        }
        float cs = 0.f;
#pragma unroll
        for (int v = 0; v < 8; ++v) {
          const float p = __expf(s[v] * scale - mrow[v]) * rl[v];
          cs += p;
          pl[(v + 8 * hl) * 32 + t * 16 + m16] = (bf16_t)p;   // C-layout -> LDS
        }
        cs += __shfl_xor(cs, 16, 32);           // combine row halves
        csum[t] = cs;
      }
      if (hl == 0) {
        avg_l[wid][j + m16]      += csum[0];
        avg_l[wid][j + 16 + m16] += csum[1];
      }
      wait_dscnt0();                            // P visible before A-layout reload
      const v16bf pa = load_a_chunks(pl + m16 * 32 + koffA);
#pragma unroll
      for (int c = 0; c < 8; ++c) {
        v16bf vf = *(const v16bf*)(Vb + (size_t)(c * 16 + m16) * S_LEN + j + koffB);
        o[c] = wmma_bf16(pa, vf, o[c]);
      }
    }

    // column sums of O (for context = mean_s(attn_out) @ Wo^T later)
#pragma unroll
    for (int c = 0; c < 8; ++c) {
      float colsum = 0.f;
#pragma unroll
      for (int v = 0; v < 8; ++v) colsum += o[c][v];
      colsum += __shfl_xor(colsum, 16, 32);
      if (hl == 0) ctx_l[wid][c * 16 + m16] += colsum;
    }
  }

  __syncthreads();
  for (int i = tid; i < S_LEN; i += 128) {
    const float s4 = avg_l[0][i] + avg_l[1][i] + avg_l[2][i] + avg_l[3][i];
    avg_out[((size_t)b * NH + h) * S_LEN + i] = s4 * (1.f / S_LEN);
  }
  for (int i = tid; i < HD; i += 128) {
    const float s4 = ctx_l[0][i] + ctx_l[1][i] + ctx_l[2][i] + ctx_l[3][i];
    ctx_acc[(size_t)b * H_DIM + h * HD + i] = s4;
  }
}

// ---------------- context = (ctx_acc/S) @ Wo^T + bo ----------------
__global__ void ctx_kernel(const float* __restrict__ ctx_acc,
                           const float* __restrict__ Wo,
                           const float* __restrict__ bo,
                           float* __restrict__ context) {
  const int idx = blockIdx.x * blockDim.x + threadIdx.x;
  if (idx >= BATCH * H_DIM) return;
  const int b = idx / H_DIM, j = idx % H_DIM;
  const float* ar = ctx_acc + (size_t)b * H_DIM;
  const float* wr = Wo + (size_t)j * H_DIM;
  float acc = 0.f;
  for (int k = 0; k < H_DIM; ++k) acc += ar[k] * wr[k];
  context[idx] = acc * (1.f / S_LEN) + bo[j];
}

// ---------------- diversity from avg_attn ----------------
__global__ __launch_bounds__(256) void div_kernel(const float* __restrict__ avg,
                                                  float* __restrict__ dout_div) {
  __shared__ float red[256];
  const int tid = threadIdx.x;                  // 256 = B*nh*nh triples
  const int b  = tid >> 6;
  const int g  = (tid >> 3) & 7;
  const int h2 = tid & 7;
  const float* ag = avg + ((size_t)b * NH + g)  * S_LEN;
  const float* ah = avg + ((size_t)b * NH + h2) * S_LEN;
  float c = 0.f;
  for (int s = 0; s < S_LEN; ++s) c += ag[s] * ah[s];
  red[tid] = (g == h2) ? 0.f : fabsf(c);
  __syncthreads();
  for (int off = 128; off > 0; off >>= 1) {
    if (tid < off) red[tid] += red[tid + off];
    __syncthreads();
  }
  if (tid == 0) *dout_div = red[0] * (1.f / 256.f);
}

// ---------------- launch ----------------
extern "C" void kernel_launch(void* const* d_in, const int* in_sizes, int n_in,
                              void* d_out, int out_size, void* d_ws, size_t ws_size,
                              hipStream_t stream) {
  const float* x  = (const float*)d_in[0];
  const float* Wq = (const float*)d_in[1];
  const float* bq = (const float*)d_in[2];
  const float* Wk = (const float*)d_in[3];
  const float* bk = (const float*)d_in[4];
  const float* Wv = (const float*)d_in[5];
  const float* bv = (const float*)d_in[6];
  const float* Wo = (const float*)d_in[7];
  const float* bo = (const float*)d_in[8];

  float* out      = (float*)d_out;
  float* context  = out;                        // [B,H]           = 4096
  float* avg      = out + BATCH * H_DIM;        // [B,nh,S]        = 65536
  float* dvr      = avg + BATCH * NH * S_LEN;   // scalar

  const size_t NX = (size_t)BATCH * S_LEN * H_DIM;  // 8388608
  const size_t NW = (size_t)H_DIM * H_DIM;          // 1048576

  char* ws = (char*)d_ws;
  size_t off = 0;
  auto take = [&](size_t bytes) -> char* {
    char* p = ws + off;
    off = (off + bytes + 255) & ~(size_t)255;
    return p;
  };
  bf16_t* xb   = (bf16_t*)take(NX * sizeof(bf16_t));
  bf16_t* wqb  = (bf16_t*)take(NW * sizeof(bf16_t));
  bf16_t* wkb  = (bf16_t*)take(NW * sizeof(bf16_t));
  bf16_t* wvb  = (bf16_t*)take(NW * sizeof(bf16_t));
  bf16_t* Qb   = (bf16_t*)take(NX * sizeof(bf16_t));
  bf16_t* Kb   = (bf16_t*)take(NX * sizeof(bf16_t));
  bf16_t* Vtb  = (bf16_t*)take(NX * sizeof(bf16_t));   // [B][H][S]
  float*  ctxa = (float*)take((size_t)BATCH * H_DIM * sizeof(float));
  (void)ws_size; (void)in_sizes; (void)n_in; (void)out_size;

  cvt_bf16_kernel<<<4096, 256, 0, stream>>>(x,  xb,  (int)NX);
  cvt_bf16_kernel<<<1024, 256, 0, stream>>>(Wq, wqb, (int)NW);
  cvt_bf16_kernel<<<1024, 256, 0, stream>>>(Wk, wkb, (int)NW);
  cvt_bf16_kernel<<<1024, 256, 0, stream>>>(Wv, wvb, (int)NW);

  // 512 M-tiles * 64 N-tiles = 32768 wave-tiles / 8 waves per block = 4096 blocks
  qkv_gemm_kernel<<<4096, 256, 0, stream>>>(xb, wqb, bq, Qb,  0);
  qkv_gemm_kernel<<<4096, 256, 0, stream>>>(xb, wkb, bk, Kb,  0);
  qkv_gemm_kernel<<<4096, 256, 0, stream>>>(xb, wvb, bv, Vtb, 1);

  attn_kernel<<<BATCH * NH, 128, 0, stream>>>(Qb, Kb, Vtb, avg, ctxa);

  ctx_kernel<<<(BATCH * H_DIM + 255) / 256, 256, 0, stream>>>(ctxa, Wo, bo, context);
  div_kernel<<<1, 256, 0, stream>>>(avg, dvr);
}
